// GATLayer_15625091023233
// MI455X (gfx1250) — matrix operand order repaired and checked by reading, not compiled, not run on hardware
//
#include <hip/hip_runtime.h>

#define N_NODES 50000
#define N_EDGES 800000
#define IN_F    128
#define HEADS   8
#define D_HEAD  16
#define OUT_F   128   // HEADS * D_HEAD

typedef __attribute__((ext_vector_type(16))) __bf16 v16bf;
typedef __attribute__((ext_vector_type(8)))  float  v8f;

// round-to-nearest-even float -> bf16
__device__ __forceinline__ __bf16 f2bf(float f) {
    unsigned u = __float_as_uint(f);
    unsigned r = u + 0x7FFFu + ((u >> 16) & 1u);
    unsigned short hs = (unsigned short)(r >> 16);
    return __builtin_bit_cast(__bf16, hs);
}

// -------------------------------------------------------------------------
// Kernel 1: h = x @ W^T  via v_wmma_f32_16x16x32_bf16
// grid.x = N_NODES/16 (=3125), block = 256 (8 waves); wave w -> out cols [16w,16w+16)
// -------------------------------------------------------------------------
__global__ __launch_bounds__(256) void gat_gemm_wmma(const float* __restrict__ x,
                                                     const float* __restrict__ W,
                                                     float* __restrict__ h) {
    const int wave = threadIdx.x >> 5;      // 0..7 : output column tile
    const int lane = threadIdx.x & 31;
    const int l16  = lane & 15;
    const int hi   = lane >> 4;             // 0: lanes 0-15, 1: lanes 16-31
    const int row0 = blockIdx.x * 16;       // node tile base
    const int ncol = wave * 16 + l16;       // output feature (B column)

    v8f acc = {};
#pragma unroll
    for (int kk = 0; kk < 4; ++kk) {
        const int kbase = kk * 32;
        v16bf a, b;
        // A-matrix 16x32 bf16 (ISA layout): lane<16 holds K {0..7,16..23} of row l16,
        // lane>=16 same rows with K offset +8.
        const float* xr = x + (size_t)(row0 + l16) * IN_F + kbase + hi * 8;
#pragma unroll
        for (int j = 0; j < 16; ++j) {
            const int kl = (j < 8) ? j : (j + 8);   // {0..7, 16..23} relative
            a[j] = f2bf(xr[kl]);
        }
        // B-matrix 32x16 bf16: lanes 0-15 hold K {0..7,16..23}, lanes 16-31 K offset +8,
        // column N = l16.  B[k][n] = W^T[k][n] = W[n][k].
        const float* wr = W + (size_t)ncol * IN_F + kbase + hi * 8;
#pragma unroll
        for (int j = 0; j < 16; ++j) b[j] = f2bf(wr[(j < 8) ? j : (j + 8)]);

        acc = __builtin_amdgcn_wmma_f32_16x16x32_bf16(false, a, false, b,
                                                      (short)0, acc, false, false);
    }
    // D layout: VGPR r -> row r (lanes 0-15) / row r+8 (lanes 16-31), col = l16
#pragma unroll
    for (int r = 0; r < 8; ++r) {
        const int row = row0 + r + hi * 8;
        h[(size_t)row * OUT_F + wave * 16 + l16] = acc[r];
    }
}

// -------------------------------------------------------------------------
// Kernel 2: per-node attention logits  s_src/s_dst [N, H]
// -------------------------------------------------------------------------
__global__ void gat_scores(const float* __restrict__ h,
                           const float* __restrict__ a_src,
                           const float* __restrict__ a_dst,
                           float* __restrict__ s_src,
                           float* __restrict__ s_dst) {
    const int i = blockIdx.x * blockDim.x + threadIdx.x;   // n*HEADS + hd
    if (i >= N_NODES * HEADS) return;
    const int n  = i >> 3;
    const int hd = i & 7;
    const float* hp = h + (size_t)n * OUT_F + hd * D_HEAD;
    const float* as = a_src + hd * D_HEAD;
    const float* ad = a_dst + hd * D_HEAD;
    float ss = 0.f, sd = 0.f;
#pragma unroll
    for (int d = 0; d < D_HEAD; ++d) {
        const float v = hp[d];
        ss += v * as[d];
        sd += v * ad[d];
    }
    s_src[i] = ss;
    s_dst[i] = sd;
}

__device__ __forceinline__ float leaky02(float a) {
    return a > 0.f ? a : 0.2f * a;
}

// -------------------------------------------------------------------------
// Kernel 3: segment max of leaky_relu logits onto zero-initialized attn_max.
// Base is 0 (reference clamps at 0), so non-negative-float == unsigned order.
// -------------------------------------------------------------------------
__global__ void gat_edge_max(const int* __restrict__ ei,
                             const float* __restrict__ s_src,
                             const float* __restrict__ s_dst,
                             float* __restrict__ attn_max) {
    const int i = blockIdx.x * blockDim.x + threadIdx.x;
    if (i >= N_EDGES * HEADS) return;
    const int e  = i >> 3;
    const int hd = i & 7;
    const int src = ei[e];
    const int dst = ei[N_EDGES + e];
    const float a = leaky02(s_src[src * HEADS + hd] + s_dst[dst * HEADS + hd]);
    if (a > 0.f)
        atomicMax((unsigned int*)(attn_max + dst * HEADS + hd), __float_as_uint(a));
}

// -------------------------------------------------------------------------
// Kernel 4: attn_sum[dst,h] += exp(attn - attn_max[dst,h])
// -------------------------------------------------------------------------
__global__ void gat_edge_sum(const int* __restrict__ ei,
                             const float* __restrict__ s_src,
                             const float* __restrict__ s_dst,
                             const float* __restrict__ attn_max,
                             float* __restrict__ attn_sum) {
    const int i = blockIdx.x * blockDim.x + threadIdx.x;
    if (i >= N_EDGES * HEADS) return;
    const int e  = i >> 3;
    const int hd = i & 7;
    const int src = ei[e];
    const int dst = ei[N_EDGES + e];
    const float a = leaky02(s_src[src * HEADS + hd] + s_dst[dst * HEADS + hd]);
    const float ex = __expf(a - attn_max[dst * HEADS + hd]);
    atomicAdd(attn_sum + dst * HEADS + hd, ex);
}

// -------------------------------------------------------------------------
// Kernel 5: out[dst] += alpha * h[src]   (alpha recomputed; h gathers hit L2)
// -------------------------------------------------------------------------
__global__ void gat_edge_scatter(const int* __restrict__ ei,
                                 const float* __restrict__ s_src,
                                 const float* __restrict__ s_dst,
                                 const float* __restrict__ attn_max,
                                 const float* __restrict__ attn_sum,
                                 const float* __restrict__ h,
                                 float* __restrict__ out) {
    const int i = blockIdx.x * blockDim.x + threadIdx.x;
    if (i >= N_EDGES * HEADS) return;
    const int e  = i >> 3;
    const int hd = i & 7;
    const int src = ei[e];
    const int dst = ei[N_EDGES + e];
    const float a = leaky02(s_src[src * HEADS + hd] + s_dst[dst * HEADS + hd]);
    const float ex = __expf(a - attn_max[dst * HEADS + hd]);
    const float alpha = ex / (attn_sum[dst * HEADS + hd] + 1e-10f);

    const float4* hp = (const float4*)(h + (size_t)src * OUT_F + hd * D_HEAD);
    float* op = out + (size_t)dst * OUT_F + hd * D_HEAD;
#pragma unroll
    for (int q = 0; q < 4; ++q) {
        const float4 v = hp[q];
        atomicAdd(op + 4 * q + 0, v.x * alpha);
        atomicAdd(op + 4 * q + 1, v.y * alpha);
        atomicAdd(op + 4 * q + 2, v.z * alpha);
        atomicAdd(op + 4 * q + 3, v.w * alpha);
    }
}

// -------------------------------------------------------------------------
extern "C" void kernel_launch(void* const* d_in, const int* in_sizes, int n_in,
                              void* d_out, int out_size, void* d_ws, size_t ws_size,
                              hipStream_t stream) {
    const float* x     = (const float*)d_in[0];
    const int*   ei    = (const int*)d_in[1];   // [2, E]: row0 = src, row1 = dst
    const float* W     = (const float*)d_in[2];
    const float* a_src = (const float*)d_in[3];
    const float* a_dst = (const float*)d_in[4];
    float* out = (float*)d_out;

    // workspace layout (all 256B-aligned by construction)
    char* ws = (char*)d_ws;
    size_t off = 0;
    float* h = (float*)(ws + off);        off += (size_t)N_NODES * OUT_F * sizeof(float);
    float* s_src = (float*)(ws + off);    off += (size_t)N_NODES * HEADS * sizeof(float);
    float* s_dst = (float*)(ws + off);    off += (size_t)N_NODES * HEADS * sizeof(float);
    float* attn_max = (float*)(ws + off); off += (size_t)N_NODES * HEADS * sizeof(float);
    float* attn_sum = (float*)(ws + off); off += (size_t)N_NODES * HEADS * sizeof(float);
    (void)ws_size; (void)in_sizes; (void)n_in;

    hipMemsetAsync(attn_max, 0, (size_t)N_NODES * HEADS * sizeof(float), stream);
    hipMemsetAsync(attn_sum, 0, (size_t)N_NODES * HEADS * sizeof(float), stream);
    hipMemsetAsync(out, 0, (size_t)out_size * sizeof(float), stream);

    gat_gemm_wmma<<<N_NODES / 16, 256, 0, stream>>>(x, W, h);

    const int nh = N_NODES * HEADS;
    gat_scores<<<(nh + 255) / 256, 256, 0, stream>>>(h, a_src, a_dst, s_src, s_dst);

    const int eh = N_EDGES * HEADS;
    gat_edge_max<<<(eh + 255) / 256, 256, 0, stream>>>(ei, s_src, s_dst, attn_max);
    gat_edge_sum<<<(eh + 255) / 256, 256, 0, stream>>>(ei, s_src, s_dst, attn_max, attn_sum);
    gat_edge_scatter<<<(eh + 255) / 256, 256, 0, stream>>>(ei, s_src, s_dst, attn_max,
                                                           attn_sum, h, out);
}
